// GroupedVectorQuantizer_81750407512158
// MI455X (gfx1250) — matrix operand order repaired and checked by reading, not compiled, not run on hardware
//
#include <hip/hip_runtime.h>

typedef __attribute__((ext_vector_type(16))) _Float16 v16h;
typedef __attribute__((ext_vector_type(8)))  float    v8f;

#define N_TOK  16384   // B*T = 8*2048
#define GROUPS 8
#define KCODES 1024
#define GD     64
#define DDIM   512
#define NTILES (KCODES / 16)

// ws region sizes (bytes). cb16 gets 2 tiles (4KB) of padding so the
// double-buffered prefetch may harmlessly read 2 tiles past the end on the
// last loop iteration (values are never consumed). cb2 gets 128B padding
// for the same reason.
#define CB16_BYTES  ((size_t)GROUPS * KCODES * GD * 2)
#define CB16_PAD    4096
#define CB2_BYTES   ((size_t)GROUPS * KCODES * 4)
#define CB2_PAD     128

// ---------------------------------------------------------------------------
// Kernel 1: store (-2 * codebook) as f16 (WMMA B operand) and ||c||^2 (f32).
// With B pre-scaled by -2 and the WMMA accumulator seeded with ||c||^2, the
// WMMA output IS the distance score (up to the argmin-invariant ||z||^2):
//   score = ||c||^2 + z . (-2c) = dist - ||z||^2
// One wave per code row (g,k): 64 elements, 2 per lane.
// ---------------------------------------------------------------------------
__global__ void cb_prep(const float* __restrict__ cb,
                        _Float16* __restrict__ cb16,
                        float* __restrict__ cb2) {
    const int waveId = (blockIdx.x * blockDim.x + threadIdx.x) >> 5; // g*K + k
    const int lane   = threadIdx.x & 31;
    if (waveId >= GROUPS * KCODES) return;
    const float* src = cb + (size_t)waveId * GD;
    _Float16*    dst = cb16 + (size_t)waveId * GD;
    float s = 0.f;
#pragma unroll
    for (int i = 0; i < 2; ++i) {
        float v = src[lane + 32 * i];
        s += v * v;
        dst[lane + 32 * i] = (_Float16)(-2.0f * v);
    }
#pragma unroll
    for (int m = 16; m >= 1; m >>= 1) s += __shfl_xor(s, m, 32);
    if (lane == 0) cb2[waveId] = s;
}

// ---------------------------------------------------------------------------
// Kernel 2: WMMA nearest-code search, 2x unrolled, double-buffered prefetch
// with linear (strength-reducible) prefetch addressing. Wave = 16 token-rows
// x 1 group; 2x v_wmma_f32_16x16x32_f16 per 16-code tile accumulate the
// gd=64 contraction on top of C = ||c||^2.
// ---------------------------------------------------------------------------
__global__ void argmin_kernel(const float* __restrict__ ze,
                              const _Float16* __restrict__ cb16,
                              const float* __restrict__ cb2,
                              int* __restrict__ idxOut) {
    const int wave = threadIdx.x >> 5;
    const int lane = threadIdx.x & 31;
    const int hf   = lane >> 4;        // which 16-lane half
    const int m    = lane & 15;
    const int rt   = blockIdx.x * 8 + wave;   // 16-row tile index
    const int g    = blockIdx.y;
    const int row  = rt * 16 + m;

    // A-operand (16x64 f32 -> two 16x32 f16 tiles), contiguous layout:
    // element j of lane maps to K = j + 16*hf  (+32 for a1)
    const float* zrow = ze + (size_t)row * DDIM + g * GD;
    v16h a0, a1;
#pragma unroll
    for (int j = 0; j < 8; ++j) {
        a0[j]     = (_Float16)zrow[      16 * hf + j];
        a0[j + 8] = (_Float16)zrow[ 8 + 16 * hf + j];
        a1[j]     = (_Float16)zrow[32 + 16 * hf + j];
        a1[j + 8] = (_Float16)zrow[40 + 16 * hf + j];
    }

    float bestS[8];
    int   bestI[8];
#pragma unroll
    for (int v = 0; v < 8; ++v) { bestS[v] = 3.4e38f; bestI[v] = 0; }

    // B-operand: element j of lane -> K(d) = j + 16*hf; column = lane&15.
    // 32B-aligned contiguous 16-half chunks; tile stride = 16*GD halfs.
    const _Float16* bbase = cb16 + ((size_t)g * KCODES + m) * GD + 16 * hf;
    const float*    c2p   = cb2 + g * KCODES + m;

    // Double-buffered pipeline: buffer A = even tiles, buffer B = odd tiles.
    v16h b0a = *(const v16h*)(bbase);
    v16h b1a = *(const v16h*)(bbase + 32);
    float c2a = c2p[0];
    v16h b0b = *(const v16h*)(bbase + 16 * GD);
    v16h b1b = *(const v16h*)(bbase + 16 * GD + 32);
    float c2b = c2p[16];

    for (int t = 0; t < NTILES; t += 2) {
        // ---- even tile t: consume buffer A, prefetch tile t+2 into A ----
        {
            v8f acc = {c2a, c2a, c2a, c2a, c2a, c2a, c2a, c2a};
            acc = __builtin_amdgcn_wmma_f32_16x16x32_f16(false, a0, false, b0a,
                                                         (short)0, acc, false, false);
            acc = __builtin_amdgcn_wmma_f32_16x16x32_f16(false, a1, false, b1a,
                                                         (short)0, acc, false, false);
            // Linear prefetch; final iteration reads into the 4KB pad region
            // (values never consumed).
            const _Float16* nb = bbase + (size_t)(t + 2) * 16 * GD;
            b0a = *(const v16h*)(nb);
            b1a = *(const v16h*)(nb + 32);
            c2a = c2p[(t + 2) * 16];

            const int code = t * 16 + m;
#pragma unroll
            for (int v = 0; v < 8; ++v) {
                bool lt = acc[v] < bestS[v];      // one v_cmp, two v_cndmask
                bestS[v] = lt ? acc[v] : bestS[v];
                bestI[v] = lt ? code   : bestI[v];
            }
        }
        // ---- odd tile t+1: consume buffer B, prefetch tile t+3 into B ----
        {
            v8f acc = {c2b, c2b, c2b, c2b, c2b, c2b, c2b, c2b};
            acc = __builtin_amdgcn_wmma_f32_16x16x32_f16(false, a0, false, b0b,
                                                         (short)0, acc, false, false);
            acc = __builtin_amdgcn_wmma_f32_16x16x32_f16(false, a1, false, b1b,
                                                         (short)0, acc, false, false);
            const _Float16* nb = bbase + (size_t)(t + 3) * 16 * GD;
            b0b = *(const v16h*)(nb);
            b1b = *(const v16h*)(nb + 32);
            c2b = c2p[(t + 3) * 16];

            const int code = (t + 1) * 16 + m;
#pragma unroll
            for (int v = 0; v < 8; ++v) {
                bool lt = acc[v] < bestS[v];
                bestS[v] = lt ? acc[v] : bestS[v];
                bestI[v] = lt ? code   : bestI[v];
            }
        }
    }

    // Reduce (min, argmin) across the 16 lanes of each half
    // (xor masks < 16 stay inside a half on wave32).
#pragma unroll
    for (int msk = 8; msk >= 1; msk >>= 1) {
#pragma unroll
        for (int v = 0; v < 8; ++v) {
            float os = __shfl_xor(bestS[v], msk, 32);
            int   oi = __shfl_xor(bestI[v], msk, 32);
            if (os < bestS[v] || (os == bestS[v] && oi < bestI[v])) {
                bestS[v] = os; bestI[v] = oi;
            }
        }
    }

    if (m == 0) {
#pragma unroll
        for (int v = 0; v < 8; ++v) {
            int r = v + 8 * hf;                // hf=0 -> rows 0..7, hf=1 -> 8..15
            idxOut[(size_t)(rt * 16 + r) * GROUPS + g] = bestI[v];
        }
    }
}

// ---------------------------------------------------------------------------
// Kernel 3: gather z_q (full f32) + deterministic per-block loss partials.
// One wave per (n,g) pair; each lane handles a contiguous float2 (b64 ld/st).
// ---------------------------------------------------------------------------
__global__ void gather_loss(const float* __restrict__ ze,
                            const float* __restrict__ cb,
                            const int* __restrict__ idx,
                            float* __restrict__ zq,
                            float* __restrict__ partials) {
    __shared__ float sdata[8];
    const int wave = threadIdx.x >> 5;
    const int lane = threadIdx.x & 31;
    const int p    = blockIdx.x * 8 + wave;    // p = n*G + g
    const int n    = p >> 3;
    const int g    = p & 7;
    const int k    = idx[p];

    const float2* crow = (const float2*)(cb + ((size_t)g * KCODES + k) * GD);
    const float2* zrow = (const float2*)(ze + (size_t)n * DDIM + g * GD);
    float2*       qrow = (float2*)(zq + (size_t)n * DDIM + g * GD);

    float2 c = crow[lane];
    float2 z = zrow[lane];
    qrow[lane] = c;                            // z_q_st forward value == z_q
    float dx = c.x - z.x;
    float dy = c.y - z.y;
    float s  = dx * dx + dy * dy;

#pragma unroll
    for (int msk = 16; msk >= 1; msk >>= 1) s += __shfl_xor(s, msk, 32);
    if (lane == 0) sdata[wave] = s;
    __syncthreads();
    if (threadIdx.x == 0) {
        float tot = 0.f;
#pragma unroll
        for (int w = 0; w < 8; ++w) tot += sdata[w];
        partials[blockIdx.x] = tot;
    }
}

// ---------------------------------------------------------------------------
// Kernel 4: deterministic final reduction.
// vq_loss = codebook_loss + commit_loss = (2 / (N*GD)) * total_sq_err
// ---------------------------------------------------------------------------
__global__ void loss_reduce(const float* __restrict__ partials,
                            float* __restrict__ lossOut, int nPart) {
    __shared__ float sdata[256];
    float s = 0.f;
    for (int i = threadIdx.x; i < nPart; i += 256) s += partials[i];
    sdata[threadIdx.x] = s;
    __syncthreads();
    for (int off = 128; off >= 1; off >>= 1) {
        if ((int)threadIdx.x < off) sdata[threadIdx.x] += sdata[threadIdx.x + off];
        __syncthreads();
    }
    if (threadIdx.x == 0)
        *lossOut = sdata[0] * (2.0f / (float)((size_t)N_TOK * GD));
}

extern "C" void kernel_launch(void* const* d_in, const int* in_sizes, int n_in,
                              void* d_out, int out_size, void* d_ws, size_t ws_size,
                              hipStream_t stream) {
    (void)in_sizes; (void)n_in; (void)out_size; (void)ws_size;

    const float* ze = (const float*)d_in[0];   // (B,T,D) f32
    const float* cb = (const float*)d_in[1];   // (G,K,gd) f32

    // d_out flat layout: z_q (N*D f32) | vq_loss (1 f32) | idx (N*G i32)
    float* zq      = (float*)d_out;
    float* lossOut = zq + (size_t)N_TOK * DDIM;
    int*   idxOut  = (int*)d_out + ((size_t)N_TOK * DDIM + 1);

    // Workspace: cb16 (1MB) | 4KB pad | cb2 (32KB) | 128B pad | partials (64KB)
    _Float16* cb16 = (_Float16*)d_ws;
    float* cb2 = (float*)((char*)d_ws + CB16_BYTES + CB16_PAD);
    float* partials = (float*)((char*)cb2 + CB2_BYTES + CB2_PAD);

    // 1) codebook (-2x) f16 copy + squared norms: 8192 waves
    cb_prep<<<(GROUPS * KCODES * 32) / 256, 256, 0, stream>>>(cb, cb16, cb2);

    // 2) WMMA argmin: 8 waves/block, each wave = 16 rows x 1 group
    dim3 grid(N_TOK / 16 / 8, GROUPS);
    argmin_kernel<<<grid, 256, 0, stream>>>(ze, cb16, cb2, idxOut);

    // 3) gather + loss partials: one wave per (n,g), 8 waves/block
    const int nPairs = N_TOK * GROUPS;
    gather_loss<<<nPairs / 8, 256, 0, stream>>>(ze, cb, idxOut, zq, partials);

    // 4) final deterministic loss reduction
    loss_reduce<<<1, 256, 0, stream>>>(partials, lossOut, nPairs / 8);
}
